// HeteroFTAGNN_v3_42554535969187
// MI455X (gfx1250) — matrix operand order repaired and checked by reading, not compiled, not run on hardware
//
#include <hip/hip_runtime.h>
#include <hip/hip_bf16.h>
#include <math.h>

typedef __attribute__((ext_vector_type(16))) _Float16 v16h;
typedef __attribute__((ext_vector_type(8)))  _Float16 v8h;
typedef __attribute__((ext_vector_type(8)))  float    v8f;
typedef __attribute__((ext_vector_type(4)))  float    v4f;

// ---------------------------------------------------------------------------
// WMMA GEMM:  C[M x 128] = f16(A[M x K]) @ f16(W[K x 128]) (+ bias)
// K compile-time in {32,64,128}; M % 64 == 0; N = 128.
// Block = 128 threads (4 waves); wave computes a 16-row x 128-col strip.
// W staged TRANSPOSED in LDS (sWt[n][k]) -> B fragment = contiguous v16h load.
// Per k-step: batch all 8 B fragments, then issue the 8-WMMA chain, so the
// ds_load_b128s run ahead of the matrix pipe instead of stalling each WMMA.
// ---------------------------------------------------------------------------
template <int K>
__global__ __launch_bounds__(128) void k_gemm_wmma(
    const float* __restrict__ A, const float* __restrict__ W,
    const float* __restrict__ bias, float* __restrict__ C, int M)
{
  __shared__ _Float16 sWt[128 * K];          // [n][k], transposed
  const int tid = threadIdx.x;
  for (int i = tid; i < K * 128; i += 128) {
    const int k = i >> 7, n = i & 127;
    sWt[n * K + k] = (_Float16)W[i];
  }
  __syncthreads();

  const int wave = tid >> 5;
  const int lane = tid & 31;
  const long row0 = ((long)blockIdx.x * 4 + wave) * 16;
  const int m   = lane & 15;
  const int hi  = lane >> 4;
  const int col = lane & 15;
  const float* Arow = A + (row0 + m) * (long)K;

  v8f acc[8] = {};
#pragma unroll
  for (int k0 = 0; k0 < K; k0 += 32) {
    v16h a;
    const int ka = k0 + (hi ? 16 : 0);
    const v4f q0 = *(const v4f*)&Arow[ka];
    const v4f q1 = *(const v4f*)&Arow[ka + 4];
    const v4f q2 = *(const v4f*)&Arow[ka + 8];
    const v4f q3 = *(const v4f*)&Arow[ka + 12];
#pragma unroll
    for (int i = 0; i < 4; ++i) {
      a[i]      = (_Float16)q0[i];
      a[i + 4]  = (_Float16)q1[i];
      a[i + 8]  = (_Float16)q2[i];
      a[i + 12] = (_Float16)q3[i];
    }
    const int kb = k0 + (hi ? 16 : 0);
    v16h bf[8];
#pragma unroll
    for (int t = 0; t < 8; ++t) bf[t] = *(const v16h*)&sWt[(t * 16 + col) * K + kb];
#pragma unroll
    for (int t = 0; t < 8; ++t)
      acc[t] = __builtin_amdgcn_wmma_f32_16x16x32_f16(
          false, a, false, bf[t], (short)0, acc[t], false, false);
  }
#pragma unroll
  for (int t = 0; t < 8; ++t) {
#pragma unroll
    for (int r = 0; r < 8; ++r) {
      const int row = hi * 8 + r;
      const int c   = t * 16 + col;
      float v = acc[t][r];
      if (bias) v += bias[c];
      C[(row0 + row) * 128 + c] = v;
    }
  }
}

static void gemm128(const float* A, const float* W, const float* bias, float* C,
                    int M, int K, hipStream_t s)
{
  if (K == 32)      k_gemm_wmma<32><<<M / 64, 128, 0, s>>>(A, W, bias, C, M);
  else if (K == 64) k_gemm_wmma<64><<<M / 64, 128, 0, s>>>(A, W, bias, C, M);
  else              k_gemm_wmma<128><<<M / 64, 128, 0, s>>>(A, W, bias, C, M);
}

// ---------------------------------------------------------------------------
// WMMA flash-attention mean pool:  outvec[c] += mean_rows(softmax(QK^T*s) V)[c]
// Grid = M/64 blocks, 128 threads (4 waves); wave w owns 16 query rows.
// K (row-major f16) and V^T (f16) are DOUBLE-BUFFERED in LDS: block j+1 is
// staged while block j is computed -> one barrier per 32-key block.
// B fragments are batched into registers before each WMMA chain.
// sPh (P C-layout -> A-layout round trip) is wave-private; same-wave DS ops
// are in-order, so it needs no barrier.
// ---------------------------------------------------------------------------
__global__ __launch_bounds__(128) void k_attn_pool_wmma(
    const float* __restrict__ Q, const float* __restrict__ Kmat,
    const float* __restrict__ V, int M, float* __restrict__ outvec)
{
  __shared__ _Float16 sKh[2][32 * 128];    // [j][c]  row-major f16
  __shared__ _Float16 sVt[2][128 * 32];    // [c][j]  transposed f16
  __shared__ _Float16 sPh[4][16 * 32];     // per-wave P [q][j]
  const int tid  = threadIdx.x;
  const int wave = tid >> 5;
  const int lane = tid & 31;
  const int m    = lane & 15;
  const int hi   = lane >> 4;
  const int col  = lane & 15;
  const long q0r = ((long)blockIdx.x * 4 + wave) * 16;
  const float scale = 0.088388347648318447f;   // 128^-0.5

  auto stage = [&](int buf, int j0) {
    for (int idx = tid; idx < 32 * 128; idx += 128) {
      const int j = idx >> 7, c = idx & 127;
      sKh[buf][idx]        = (_Float16)Kmat[(long)(j0 + j) * 128 + c];
      sVt[buf][c * 32 + j] = (_Float16)V[(long)(j0 + j) * 128 + c];
    }
  };

  // Q fragments (pre-scaled), contraction dim c=128 -> 4 k-steps
  v16h aq[4];
  {
    const float* Qrow = Q + (q0r + m) * 128;
#pragma unroll
    for (int kc = 0; kc < 4; ++kc) {
      const int ka = kc * 32 + (hi ? 16 : 0);
#pragma unroll
      for (int i = 0; i < 8; ++i) {
        aq[kc][i]     = (_Float16)(Qrow[ka + i] * scale);
        aq[kc][i + 8] = (_Float16)(Qrow[ka + 8 + i] * scale);
      }
    }
  }

  v8f o[8] = {};
  float mrow[8], lrow[8];
#pragma unroll
  for (int r = 0; r < 8; ++r) { mrow[r] = -3.0e38f; lrow[r] = 0.0f; }

  int cur = 0;
  stage(0, 0);
  __syncthreads();

  const int kbB = hi ? 16 : 0;
  for (int j0 = 0; j0 < M; j0 += 32) {
    if (j0 + 32 < M) stage(cur ^ 1, j0 + 32);   // overlap staging with compute

    // ---- S = (Q*scale) @ K^T : batch 8 B fragments, then 8 WMMAs ----
    v16h bk[8];
#pragma unroll
    for (int kc = 0; kc < 4; ++kc) {
      const int kb = kc * 32 + kbB;
      bk[kc * 2]     = *(const v16h*)&sKh[cur][col * 128 + kb];
      bk[kc * 2 + 1] = *(const v16h*)&sKh[cur][(16 + col) * 128 + kb];
    }
    v8f s0 = {}, s1 = {};
#pragma unroll
    for (int kc = 0; kc < 4; ++kc) {
      s0 = __builtin_amdgcn_wmma_f32_16x16x32_f16(false, aq[kc], false, bk[kc * 2],     (short)0, s0, false, false);
      s1 = __builtin_amdgcn_wmma_f32_16x16x32_f16(false, aq[kc], false, bk[kc * 2 + 1], (short)0, s1, false, false);
    }

    // ---- batch V^T fragments early (independent of the stats below) ----
    v16h bv[8];
#pragma unroll
    for (int t = 0; t < 8; ++t) bv[t] = *(const v16h*)&sVt[cur][(t * 16 + col) * 32 + kbB];

    // ---- online softmax update per row q = hi*8 + r ----
#pragma unroll
    for (int r = 0; r < 8; ++r) {
      float mt = fmaxf(s0[r], s1[r]);
#pragma unroll
      for (int off = 1; off < 16; off <<= 1) mt = fmaxf(mt, __shfl_xor(mt, off, 16));
      const float mn  = fmaxf(mrow[r], mt);
      const float fac = expf(mrow[r] - mn);
      const float p0  = expf(s0[r] - mn);
      const float p1  = expf(s1[r] - mn);
      float ps = p0 + p1;
#pragma unroll
      for (int off = 1; off < 16; off <<= 1) ps += __shfl_xor(ps, off, 16);
      lrow[r] = lrow[r] * fac + ps;
      mrow[r] = mn;
#pragma unroll
      for (int t = 0; t < 8; ++t) o[t][r] *= fac;
      sPh[wave][(hi * 8 + r) * 32 + col]      = (_Float16)p0;
      sPh[wave][(hi * 8 + r) * 32 + 16 + col] = (_Float16)p1;
    }

    // ---- P: C-layout -> A-layout via wave-private LDS (in-order DS) ----
    v16h ap;
    {
      const _Float16* pr = &sPh[wave][m * 32];
      const int k0a = hi ? 16 : 0;
      const v8h lo  = *(const v8h*)&pr[k0a];
      const v8h hi8 = *(const v8h*)&pr[8 + k0a];
#pragma unroll
      for (int i = 0; i < 8; ++i) { ap[i] = lo[i]; ap[i + 8] = hi8[i]; }
    }
    // ---- O += P @ V ----
#pragma unroll
    for (int t = 0; t < 8; ++t)
      o[t] = __builtin_amdgcn_wmma_f32_16x16x32_f16(false, ap, false, bv[t], (short)0, o[t], false, false);

    __syncthreads();   // buf[cur] free for restaging; buf[cur^1] fully staged
    cur ^= 1;
  }

  // fold rows into the mean vector: 2 atomics per column per wave
  const float invM = 1.0f / (float)M;
#pragma unroll
  for (int t = 0; t < 8; ++t) {
    float s = 0.0f;
#pragma unroll
    for (int r = 0; r < 8; ++r) s += o[t][r] / lrow[r];
    atomicAdd(&outvec[t * 16 + col], s * invM);
  }
}

// --------------------------- small helper kernels ---------------------------
__device__ __forceinline__ unsigned f2key(float f) {
  unsigned u = __float_as_uint(f);
  return (u & 0x80000000u) ? ~u : (u | 0x80000000u);
}
__device__ __forceinline__ float key2f(unsigned k) {
  return __uint_as_float((k & 0x80000000u) ? (k ^ 0x80000000u) : ~k);
}
__device__ __forceinline__ float lrelu02(float x) { return x > 0.0f ? x : 0.2f * x; }

// wa[d][h] = sum_c W[d, h*64+c] * att[h, c]
__global__ void k_make_wa(const float* __restrict__ W, const float* __restrict__ att,
                          float* __restrict__ wa, int K)
{
  int i = threadIdx.x;                 // K*2 <= 256
  if (i >= K * 2) return;
  const int d = i >> 1, h = i & 1;
  float s = 0.0f;
  for (int c = 0; c < 64; ++c) s += W[d * 128 + h * 64 + c] * att[h * 64 + c];
  wa[i] = s;
}

// out[n][h] = X[n,:] . wa[:,h]
__global__ void k_gemv2(const float* __restrict__ X, const float* __restrict__ wa,
                        float* __restrict__ out, int N, int K)
{
  long i = (long)blockIdx.x * blockDim.x + threadIdx.x;
  if (i >= (long)N * 2) return;
  const int n = (int)(i >> 1), h = (int)(i & 1);
  const float* x = X + (long)n * K;
  float s = 0.0f;
  for (int d = 0; d < K; ++d) s += x[d] * wa[d * 2 + h];
  out[i] = s;
}

__global__ void k_nan_copy(const float* __restrict__ in, float* __restrict__ out, int n)
{
  int i = blockIdx.x * blockDim.x + threadIdx.x;
  if (i >= n) return;
  float v = in[i];
  if (v != v) v = 0.0f;
  v = fminf(fmaxf(v, -3.402823466e38f), 3.402823466e38f);
  out[i] = v;
}

__global__ void k_edge_max(const int* __restrict__ src, const int* __restrict__ dst, int E,
                           const float* __restrict__ a_s, const float* __restrict__ a_d,
                           unsigned* __restrict__ mk)
{
  long i = (long)blockIdx.x * blockDim.x + threadIdx.x;
  if (i >= (long)E * 2) return;
  const int e = (int)(i >> 1), h = (int)(i & 1);
  const int sN = src[e], dN = dst[e];
  const float al = lrelu02(a_s[sN * 2 + h] + a_d[dN * 2 + h]);
  atomicMax(&mk[dN * 2 + h], f2key(al));
}

__global__ void k_edge_sumexp(const int* __restrict__ src, const int* __restrict__ dst, int E,
                              const float* __restrict__ a_s, const float* __restrict__ a_d,
                              const unsigned* __restrict__ mk, float* __restrict__ ss)
{
  long i = (long)blockIdx.x * blockDim.x + threadIdx.x;
  if (i >= (long)E * 2) return;
  const int e = (int)(i >> 1), h = (int)(i & 1);
  const int sN = src[e], dN = dst[e];
  const float al = lrelu02(a_s[sN * 2 + h] + a_d[dN * 2 + h]);
  atomicAdd(&ss[dN * 2 + h], expf(al - key2f(mk[dN * 2 + h])));
}

// coef[e][h] = softmax coefficient (computed ONCE per edge/head)
__global__ void k_edge_coef(const int* __restrict__ src, const int* __restrict__ dst, int E,
                            const float* __restrict__ a_s, const float* __restrict__ a_d,
                            const unsigned* __restrict__ mk, const float* __restrict__ ss,
                            float* __restrict__ coef)
{
  long i = (long)blockIdx.x * blockDim.x + threadIdx.x;
  if (i >= (long)E * 2) return;
  const int e = (int)(i >> 1), h = (int)(i & 1);
  const int sN = src[e], dN = dst[e];
  const float al = lrelu02(a_s[sN * 2 + h] + a_d[dN * 2 + h]);
  coef[i] = expf(al - key2f(mk[dN * 2 + h])) / (ss[dN * 2 + h] + 1e-16f);
}

// pure data movement: one wave per edge, float4 gather + 4 atomics per lane
__global__ void k_edge_aggr(const int* __restrict__ src, const int* __restrict__ dst, int E,
                            const float* __restrict__ coef,
                            const float* __restrict__ hs, float* __restrict__ out)
{
  long i = (long)blockIdx.x * blockDim.x + threadIdx.x;
  if (i >= (long)E * 32) return;
  const int e  = (int)(i >> 5);
  const int j0 = (int)(i & 31) * 4;
  const int h  = j0 >> 6;
  const int sN = src[e], dN = dst[e];
  const float a = coef[e * 2 + h];
  const v4f hv = *(const v4f*)&hs[(long)sN * 128 + j0];
  float* op = &out[(long)dN * 128 + j0];
  atomicAdd(op + 0, hv[0] * a);
  atomicAdd(op + 1, hv[1] * a);
  atomicAdd(op + 2, hv[2] * a);
  atomicAdd(op + 3, hv[3] * a);
}

// row-wise: dst = ELU(LayerNorm(acc + b0(+b1)(+b2) (+resid)) * g + beta)
__global__ __launch_bounds__(128) void k_ln_elu(
    const float* __restrict__ acc,
    const float* __restrict__ b0, const float* __restrict__ b1, const float* __restrict__ b2,
    const float* __restrict__ resid,
    const float* __restrict__ g, const float* __restrict__ beta,
    float* __restrict__ dst)
{
  const long row = blockIdx.x;
  const int c = threadIdx.x;
  float v = acc[row * 128 + c];
  if (b0) v += b0[c];
  if (b1) v += b1[c];
  if (b2) v += b2[c];
  if (resid) v += resid[row * 128 + c];

  __shared__ float red[4];
  float t = v;
  for (int o = 16; o; o >>= 1) t += __shfl_down(t, o, 32);
  if ((c & 31) == 0) red[c >> 5] = t;
  __syncthreads();
  const float mu = (red[0] + red[1] + red[2] + red[3]) * (1.0f / 128.0f);
  __syncthreads();
  const float d = v - mu;
  t = d * d;
  for (int o = 16; o; o >>= 1) t += __shfl_down(t, o, 32);
  if ((c & 31) == 0) red[c >> 5] = t;
  __syncthreads();
  const float var = (red[0] + red[1] + red[2] + red[3]) * (1.0f / 128.0f);
  float y = d * rsqrtf(var + 1e-5f) * g[c] + beta[c];
  y = (y > 0.0f) ? y : (expf(y) - 1.0f);
  dst[row * 128 + c] = y;
}

__global__ void k_scale_rows(const float* __restrict__ x, const float* __restrict__ pv,
                             float* __restrict__ out, int M)
{
  long i = (long)blockIdx.x * blockDim.x + threadIdx.x;
  if (i >= (long)M * 128) return;
  out[i] = x[i] * pv[i >> 7];
}

__global__ __launch_bounds__(128) void k_mean_rows(const float* __restrict__ X,
                                                   float* __restrict__ out, int M)
{
  const int c = threadIdx.x;
  float s = 0.0f;
  for (long r = blockIdx.x; r < M; r += gridDim.x) s += X[r * 128 + c];
  atomicAdd(&out[c], s * (1.0f / (float)M));
}

__global__ __launch_bounds__(64) void k_head(
    const float* __restrict__ lv, const float* __restrict__ gv,
    const float* __restrict__ sp, const float* __restrict__ gf,
    const float* __restrict__ W1, const float* __restrict__ b1,
    const float* __restrict__ W2, const float* __restrict__ b2,
    float* __restrict__ out)
{
  __shared__ float h[416];
  __shared__ float z[64];
  const int tid = threadIdx.x;
  for (int i = tid; i < 128; i += 64) { h[i] = lv[i]; h[128 + i] = gv[i]; h[256 + i] = sp[i]; }
  if (tid < 32) h[384 + tid] = gf[tid];
  __syncthreads();
  float s = 0.0f;
  for (int d = 0; d < 416; ++d) s += h[d] * W1[d * 64 + tid];
  s += b1[tid];
  z[tid] = s > 0.0f ? s : 0.0f;
  __syncthreads();
  if (tid == 0) {
    float o = b2[0];
    for (int j = 0; j < 64; ++j) o += z[j] * W2[j];
    out[0] = o;
  }
}

// --------------------------------- host side --------------------------------
struct GatBufs {
  float* hs; float* a_s; float* a_d;
  unsigned* mk; float* ss; float* wa_s; float* wa_d; float* coef;
};

static void run_gat(const float* Ws, const float* Wd,
                    const float* att_s, const float* att_d,
                    const float* xsrc, int Nsrc, int Ksrc,
                    const float* xdst, int Ndst, int Kdst,
                    const int* eidx, int E,
                    const GatBufs& gb, float* outacc, hipStream_t s)
{
  gemm128(xsrc, Ws, nullptr, gb.hs, Nsrc, Ksrc, s);
  k_make_wa<<<1, 256, 0, s>>>(Ws, att_s, gb.wa_s, Ksrc);
  k_make_wa<<<1, 256, 0, s>>>(Wd, att_d, gb.wa_d, Kdst);
  k_gemv2<<<(Nsrc * 2 + 255) / 256, 256, 0, s>>>(xsrc, gb.wa_s, gb.a_s, Nsrc, Ksrc);
  k_gemv2<<<(Ndst * 2 + 255) / 256, 256, 0, s>>>(xdst, gb.wa_d, gb.a_d, Ndst, Kdst);
  hipMemsetAsync(gb.mk, 0, sizeof(unsigned) * (size_t)Ndst * 2, s);
  hipMemsetAsync(gb.ss, 0, sizeof(float) * (size_t)Ndst * 2, s);
  const int* src = eidx;
  const int* dst = eidx + E;
  const long n2 = (long)E * 2;
  k_edge_max<<<(int)((n2 + 255) / 256), 256, 0, s>>>(src, dst, E, gb.a_s, gb.a_d, gb.mk);
  k_edge_sumexp<<<(int)((n2 + 255) / 256), 256, 0, s>>>(src, dst, E, gb.a_s, gb.a_d, gb.mk, gb.ss);
  k_edge_coef<<<(int)((n2 + 255) / 256), 256, 0, s>>>(src, dst, E, gb.a_s, gb.a_d, gb.mk, gb.ss, gb.coef);
  const long n32 = (long)E * 32;
  k_edge_aggr<<<(int)((n32 + 255) / 256), 256, 0, s>>>(src, dst, E, gb.coef, gb.hs, outacc);
}

extern "C" void kernel_launch(void* const* d_in, const int* in_sizes, int n_in,
                              void* d_out, int out_size, void* d_ws, size_t ws_size,
                              hipStream_t stream)
{
  (void)n_in; (void)out_size; (void)ws_size;
  const int NF = 8192, T = 2048, NS = 65536;

  // Flattened input indexing (recursive dict-insertion order of setup_inputs()):
  const float* x_fm = (const float*)d_in[0];
  const float* x_sm = (const float*)d_in[1];
  const float* gf   = (const float*)d_in[2];
  const float* pvol = (const float*)d_in[3];
  const float* emb  = (const float*)d_in[4];
#define IN(i) ((const float*)d_in[(i)])
  // conv1: qoq 5-8 | belongs 9-13 | crit_period 14-18 | crit_day 19-23 | rev_belongs 24-28
  // conv2: qoq 29-32 | belongs 33-36 | crit_period 37-40 | crit_day 41-44 | rev_belongs 45-48
  // ln: fm g49 b50 | tp g51 b52 | sm g53 b54
  // gatt 55-60 (Wq bq Wk bk Wv bv) | latt 61-66 | head 67-70 (W1 b1 W2 b2)
  const int* e_qoq = (const int*)d_in[71]; const int E_qoq = in_sizes[71] / 2;
  const int* e_bel = (const int*)d_in[72]; const int E_bel = in_sizes[72] / 2;
  const int* e_cp  = (const int*)d_in[73]; const int E_cp  = in_sizes[73] / 2;
  const int* e_cd  = (const int*)d_in[74]; const int E_cd  = in_sizes[74] / 2;
  const int* e_rb  = (const int*)d_in[75]; const int E_rb  = in_sizes[75] / 2;
  int Emax = E_qoq;
  if (E_bel > Emax) Emax = E_bel;
  if (E_cp  > Emax) Emax = E_cp;
  if (E_cd  > Emax) Emax = E_cd;
  if (E_rb  > Emax) Emax = E_rb;

  // Workspace carve (~101 MB total).
  char* base = (char*)d_ws;
  size_t off = 0;
  auto carve = [&](size_t nfloats) -> float* {
    float* p = (float*)(base + off);
    off += ((nfloats * 4) + 255) & ~(size_t)255;
    return p;
  };
  float* FM1 = carve((size_t)NF * 128);
  float* TP1 = carve((size_t)T * 128);
  float* SM1 = carve((size_t)NS * 128);
  float* FM2 = carve((size_t)NF * 128);
  float* TP2 = carve((size_t)T * 128);
  float* SM2 = carve((size_t)NS * 128);
  float* HS  = carve((size_t)NF * 128);
  float* AS  = carve((size_t)NF * 2);
  float* AD  = carve((size_t)NS * 2);
  unsigned* MK = (unsigned*)carve((size_t)NS * 2);
  float* SS  = carve((size_t)NS * 2);
  float* CO  = carve((size_t)Emax * 2);
  float* XT  = carve((size_t)T * 64);
  float* WAS = carve(256);
  float* WAD = carve(256);
  float* TPS = carve((size_t)T * 128);
  float* QB  = carve((size_t)NF * 128);
  float* KB  = carve((size_t)NF * 128);
  float* VB  = carve((size_t)NF * 128);
  float* GV  = carve(128);
  float* LV  = carve(128);
  float* SP  = carve(128);

  GatBufs gb{HS, AS, AD, MK, SS, WAS, WAD, CO};

  // pe = nan_to_num(emb_table)
  k_nan_copy<<<(T * 64 + 255) / 256, 256, 0, stream>>>(emb, XT, T * 64);

  // ---------------- conv1 ----------------
  hipMemsetAsync(FM1, 0, (size_t)NF * 128 * 4, stream);
  hipMemsetAsync(TP1, 0, (size_t)T * 128 * 4, stream);
  hipMemsetAsync(SM1, 0, (size_t)NS * 128 * 4, stream);
  run_gat(IN(5),  IN(5),  IN(6),  IN(7),  x_fm, NF, 64, x_fm, NF, 64, e_qoq, E_qoq, gb, FM1, stream);
  run_gat(IN(14), IN(15), IN(16), IN(17), XT,   T,  64, x_fm, NF, 64, e_cp,  E_cp,  gb, FM1, stream);
  run_gat(IN(24), IN(25), IN(26), IN(27), XT,   T,  64, x_fm, NF, 64, e_rb,  E_rb,  gb, FM1, stream);
  run_gat(IN(9),  IN(10), IN(11), IN(12), x_fm, NF, 64, XT,   T,  64, e_bel, E_bel, gb, TP1, stream);
  run_gat(IN(19), IN(20), IN(21), IN(22), XT,   T,  64, x_sm, NS, 32, e_cd,  E_cd,  gb, SM1, stream);

  k_ln_elu<<<NF, 128, 0, stream>>>(FM1, IN(8), IN(18), IN(28), nullptr, IN(49), IN(50), FM1);
  k_ln_elu<<<T,  128, 0, stream>>>(TP1, IN(13), nullptr, nullptr, nullptr, IN(51), IN(52), TP1);
  k_ln_elu<<<NS, 128, 0, stream>>>(SM1, IN(23), nullptr, nullptr, nullptr, IN(53), IN(54), SM1);

  // ---------------- conv2 ----------------
  hipMemsetAsync(FM2, 0, (size_t)NF * 128 * 4, stream);
  hipMemsetAsync(TP2, 0, (size_t)T * 128 * 4, stream);
  hipMemsetAsync(SM2, 0, (size_t)NS * 128 * 4, stream);
  run_gat(IN(29), IN(29), IN(30), IN(31), FM1, NF, 128, FM1, NF, 128, e_qoq, E_qoq, gb, FM2, stream);
  run_gat(IN(37), IN(37), IN(38), IN(39), TP1, T,  128, FM1, NF, 128, e_cp,  E_cp,  gb, FM2, stream);
  run_gat(IN(45), IN(45), IN(46), IN(47), TP1, T,  128, FM1, NF, 128, e_rb,  E_rb,  gb, FM2, stream);
  run_gat(IN(33), IN(33), IN(34), IN(35), FM1, NF, 128, TP1, T,  128, e_bel, E_bel, gb, TP2, stream);
  run_gat(IN(41), IN(41), IN(42), IN(43), TP1, T,  128, SM1, NS, 128, e_cd,  E_cd,  gb, SM2, stream);

  k_ln_elu<<<NF, 128, 0, stream>>>(FM2, IN(32), IN(40), IN(48), FM1, IN(49), IN(50), FM2);
  k_ln_elu<<<T,  128, 0, stream>>>(TP2, IN(36), nullptr, nullptr, TP1, IN(51), IN(52), TP2);
  k_ln_elu<<<NS, 128, 0, stream>>>(SM2, IN(44), nullptr, nullptr, SM1, IN(53), IN(54), SM2);

  // ---------------- SDPA pools (WMMA flash attention) ----------------
  k_scale_rows<<<(T * 128 + 255) / 256, 256, 0, stream>>>(TP2, pvol, TPS, T);
  gemm128(TPS, IN(55), IN(56), QB, T, 128, stream);
  gemm128(TPS, IN(57), IN(58), KB, T, 128, stream);
  gemm128(TPS, IN(59), IN(60), VB, T, 128, stream);
  hipMemsetAsync(GV, 0, 128 * 4, stream);
  k_attn_pool_wmma<<<T / 64, 128, 0, stream>>>(QB, KB, VB, T, GV);

  gemm128(FM2, IN(61), IN(62), QB, NF, 128, stream);
  gemm128(FM2, IN(63), IN(64), KB, NF, 128, stream);
  gemm128(FM2, IN(65), IN(66), VB, NF, 128, stream);
  hipMemsetAsync(LV, 0, 128 * 4, stream);
  k_attn_pool_wmma<<<NF / 64, 128, 0, stream>>>(QB, KB, VB, NF, LV);

  hipMemsetAsync(SP, 0, 128 * 4, stream);
  k_mean_rows<<<256, 128, 0, stream>>>(SM2, SP, NS);

  // ---------------- head ----------------
  k_head<<<1, 64, 0, stream>>>(LV, GV, SP, gf, IN(67), IN(68), IN(69), IN(70), (float*)d_out);
#undef IN
}
